// TreeLSTMBranch_55001351192754
// MI455X (gfx1250) — matrix operand order repaired and checked by reading, not compiled, not run on hardware
//
#include <hip/hip_runtime.h>
#include <hip/hip_bf16.h>
#include <cstdint>
#include <cstddef>

// ---------------------------------------------------------------------------
// MI455X (gfx1250) TreeLSTM branch scoring.
//  Phase 1: bf16 WMMA GEMM for pre-activations  (memory-bound, ~205MB write)
//  Phase 2: two sequential LSTM chain kernels, U weights register-resident,
//           per-step matvec via v_wmma_f32_16x16x32_bf16 (wave32, one WGP)
//  Phase 3: tiny candidate-scoring reductions.
// ---------------------------------------------------------------------------

typedef __attribute__((ext_vector_type(16))) __bf16 v16bf;
typedef __attribute__((ext_vector_type(8)))  __bf16 v8bf;
typedef __attribute__((ext_vector_type(8)))  float  v8f;

#define HDIM 256      // hidden
#define XDIM 128      // feature
#define GDIM 1024     // 3H (iou) + H (f)
#define MU_C 0.5f

static __device__ __forceinline__ v8f wmma_bf16(v16bf a, v16bf b, v8f c) {
  // D = A(16x32,bf16) x B(32x16,bf16) + C(16x16,f32)
  return __builtin_amdgcn_wmma_f32_16x16x32_bf16(
      /*neg_a=*/false, a, /*neg_b=*/false, b,
      /*c_mod=*/(short)0, c, /*reuse_a=*/false, /*reuse_b=*/false);
}
static __device__ __forceinline__ float sigm(float x) {
  return 1.0f / (1.0f + __expf(-x));
}

// ---------------------------------------------------------------------------
// Kernel 0: pack weights to bf16, build combined bias, convert features.
//   Wc (1024x128) = [W_iou_w ; W_f_w]    (row = output gate index)
//   Uc (1024x256) = [U_iou_w ; U_f_w]
//   biasc[j] = b_iou[j] (j<768)  else W_f_b + U_f_b + b_f   (constant per col)
// ---------------------------------------------------------------------------
__global__ void prep_kernel(const float* __restrict__ feat,
                            const float* __restrict__ W_iou,
                            const float* __restrict__ W_f,
                            const float* __restrict__ U_iou,
                            const float* __restrict__ U_f,
                            const float* __restrict__ b_iou,
                            const float* __restrict__ W_f_b,
                            const float* __restrict__ U_f_b,
                            const float* __restrict__ b_f,
                            __bf16* __restrict__ Wc,
                            __bf16* __restrict__ Uc,
                            float* __restrict__ biasc,
                            __bf16* __restrict__ featb, int n) {
  long stride = (long)gridDim.x * blockDim.x;
  long tid = (long)blockIdx.x * blockDim.x + threadIdx.x;
  for (long i = tid; i < GDIM * XDIM; i += stride) {
    int r = (int)(i >> 7), k = (int)(i & 127);
    float v = (r < 768) ? W_iou[i] : W_f[(size_t)(r - 768) * XDIM + k];
    Wc[i] = (__bf16)v;
  }
  for (long i = tid; i < GDIM * HDIM; i += stride) {
    int r = (int)(i >> 8), k = (int)(i & 255);
    float v = (r < 768) ? U_iou[i] : U_f[(size_t)(r - 768) * HDIM + k];
    Uc[i] = (__bf16)v;
  }
  for (long i = tid; i < GDIM; i += stride) {
    biasc[i] = (i < 768) ? b_iou[i]
                         : (W_f_b[i - 768] + U_f_b[i - 768] + b_f[i - 768]);
  }
  for (long i = tid; i < (long)n * XDIM; i += stride)
    featb[i] = (__bf16)feat[i];
}

// ---------------------------------------------------------------------------
// Kernel 1: pre(N,1024) = featb(N,128) @ Wc^T + biasc, stored bf16.
// Block = 256 threads (8 waves). Waves tile 4(row) x 2(col); each wave
// computes a 16x64 tile = 4 N-tiles, K-loop of 4 (K=32 per WMMA).
// A lane layout (16-bit A 16x32): row = lane&15; elems 0..7 -> K=half*8..,
// elems 8..15 -> K=16+half*8.. . B lane layout (32x16): col = lane&15,
// same K order as A (two 16B loads along Wc rows).
// ---------------------------------------------------------------------------
__global__ __launch_bounds__(256) void pre_gemm_kernel(
    const __bf16* __restrict__ featb, const __bf16* __restrict__ Wc,
    const float* __restrict__ biasc, __bf16* __restrict__ pre, int n) {
  int tid = threadIdx.x, lane = tid & 31, wave = tid >> 5;
  int r = lane & 15, half = lane >> 4;
  int rowblk = blockIdx.y * 64 + (wave & 3) * 16;
  int colblk = blockIdx.x * 128 + (wave >> 2) * 64;
  int arow = rowblk + r;
  const __bf16* Ap = featb + (size_t)arow * XDIM;

  v8f acc[4] = {};
#pragma unroll
  for (int kt = 0; kt < 4; kt++) {
    int kb = kt * 32;
    v16bf a = {};
    if (arow < n) {
      v8bf lo = *(const v8bf*)(Ap + kb + half * 8);
      v8bf hi = *(const v8bf*)(Ap + kb + 16 + half * 8);
      a = __builtin_shufflevector(lo, hi, 0, 1, 2, 3, 4, 5, 6, 7, 8, 9, 10,
                                  11, 12, 13, 14, 15);
    }
#pragma unroll
    for (int nt = 0; nt < 4; nt++) {
      int col = colblk + nt * 16 + r;
      const __bf16* Bp = Wc + (size_t)col * XDIM + kb;
      v8bf blo = *(const v8bf*)(Bp + half * 8);
      v8bf bhi = *(const v8bf*)(Bp + 16 + half * 8);
      v16bf b = __builtin_shufflevector(blo, bhi, 0, 1, 2, 3, 4, 5, 6, 7, 8, 9,
                                        10, 11, 12, 13, 14, 15);
      acc[nt] = wmma_bf16(a, b, acc[nt]);
    }
  }
  // D layout: VGPR v, lanes 0-15 -> M=v, lanes 16-31 -> M=v+8; N = lane&15.
#pragma unroll
  for (int nt = 0; nt < 4; nt++) {
    int col = colblk + nt * 16 + r;
    float bs = biasc[col];
#pragma unroll
    for (int v = 0; v < 8; v++) {
      int row = rowblk + v + 8 * half;
      if (row < n) pre[(size_t)row * GDIM + col] = (__bf16)(acc[nt][v] + bs);
    }
  }
}

// ---------------------------------------------------------------------------
// Kernel 2/3: sequential LSTM chain. Single 1024-thread workgroup (32 waves,
// one WGP). Wave w holds Uc rows [32w,32w+32) in VGPRs as 2 Mtiles x 8 Ktiles
// of bf16 A operands (128 VGPRs/lane). Per step: build B by replicating the
// bf16 h vector into all 16 columns (two 16B ds reads per Ktile), 16 WMMAs,
// extract column via predicated ds_store, then 256 lanes do gate math.
// DIR=0: forward pass (stores iou_msg / c of the final node only).
// DIR=1: reverse pass (emits score_node[idx] = h . lin_w + lin_b per step).
// ---------------------------------------------------------------------------
template <int DIR>
__global__ __launch_bounds__(1024) void chain_kernel(
    const __bf16* __restrict__ pre, const __bf16* __restrict__ Uc,
    const float* __restrict__ seed_iou, const float* __restrict__ seed_c,
    float* __restrict__ iou_last, float* __restrict__ c_last,
    float* __restrict__ score, const float* __restrict__ lin_w,
    const float* __restrict__ lin_b, int n) {
  __shared__ __align__(64) __bf16 s_hb[HDIM];  // h as bf16 (WMMA B operand)
  __shared__ float s_gates[GDIM];              // h_prev @ Uc^T
  __shared__ float s_c[HDIM];                  // cell state carry
  __shared__ float s_red[8];

  int tid = threadIdx.x, lane = tid & 31, wave = tid >> 5;
  int r = lane & 15, half = lane >> 4;
  int rowbase = wave * 32;

  // Load register-resident A tiles of Uc (bf16, 16x32 each).
  v16bf aU[2][8];
#pragma unroll
  for (int mt = 0; mt < 2; mt++) {
#pragma unroll
    for (int kt = 0; kt < 8; kt++) {
      const __bf16* p = Uc + (size_t)(rowbase + mt * 16 + r) * HDIM + kt * 32;
      v8bf lo = *(const v8bf*)(p + half * 8);
      v8bf hi = *(const v8bf*)(p + 16 + half * 8);
      aU[mt][kt] = __builtin_shufflevector(lo, hi, 0, 1, 2, 3, 4, 5, 6, 7, 8,
                                           9, 10, 11, 12, 13, 14, 15);
    }
  }
  float lw = (DIR && tid < HDIM) ? lin_w[tid] : 0.0f;
  float lb = DIR ? lin_b[0] : 0.0f;

  for (int t = 0; t < n; t++) {
    int idx = DIR ? (n - 1 - t) : t;
    if (t > 0) {
      v8f acc0 = {}, acc1 = {};
#pragma unroll
      for (int kt = 0; kt < 8; kt++) {
        // B replicated across all 16 columns: every lane in a half loads the
        // same 16 bf16 of h -> D identical across N, easy extraction.
        v8bf blo = *(const v8bf*)(s_hb + kt * 32 + half * 8);
        v8bf bhi = *(const v8bf*)(s_hb + kt * 32 + 16 + half * 8);
        v16bf b = __builtin_shufflevector(blo, bhi, 0, 1, 2, 3, 4, 5, 6, 7, 8,
                                          9, 10, 11, 12, 13, 14, 15);
        acc0 = wmma_bf16(aU[0][kt], b, acc0);
        acc1 = wmma_bf16(aU[1][kt], b, acc1);
      }
#pragma unroll
      for (int v = 0; v < 8; v++) {
        if (r == v) {  // lanes v and 16+v write rows v+8*half of each Mtile
          s_gates[rowbase + v + 8 * half] = acc0[v];
          s_gates[rowbase + 16 + v + 8 * half] = acc1[v];
        }
      }
      // Prefetch the pre row we will need ~6 steps from now (2KB = 16 lines).
      int pf = DIR ? idx - 6 : idx + 6;
      if (wave == 0 && lane < 16 && pf >= 0 && pf < n)
        __builtin_prefetch(pre + (size_t)pf * GDIM + lane * 64, 0, 0);
    }
    __syncthreads();

    float sc = 0.0f;
    if (tid < HDIM) {
      int j = tid;
      const __bf16* prow = pre + (size_t)idx * GDIM;
      float pi = (float)prow[j];
      float po = (float)prow[HDIM + j];
      float pu = (float)prow[2 * HDIM + j];
      float gi, go, gu, ca;
      if (t == 0) {
        gi = pi + seed_iou[j];
        go = po + seed_iou[HDIM + j];
        gu = pu + seed_iou[2 * HDIM + j];
        ca = seed_c[j];  // node 0: c_agg = c[0], no forget gate
      } else {
        gi = pi + s_gates[j];
        go = po + s_gates[HDIM + j];
        gu = pu + s_gates[2 * HDIM + j];
        float gf = (float)prow[3 * HDIM + j] + s_gates[3 * HDIM + j];
        ca = sigm(gf) * s_c[j];
      }
      float cn = sigm(gi) * tanhf(gu) + ca;
      float hn = sigm(go) * tanhf(cn);
      s_c[j] = cn;
      s_hb[j] = (__bf16)hn;
      if (DIR) sc = hn * lw;
    }
    if (DIR) {
#pragma unroll
      for (int off = 16; off > 0; off >>= 1) sc += __shfl_xor(sc, off, 32);
      if (tid < HDIM && lane == 0) s_red[wave] = sc;
      __syncthreads();
      if (tid == 0) {
        float s = 0.0f;
#pragma unroll
        for (int w = 0; w < 8; w++) s += s_red[w];
        score[idx] = s + lb;
      }
    }
    __syncthreads();
  }

  if (!DIR) {
    // Only the FINAL node's iou_msg and c are consumed by the reverse pass.
    if (n > 1) {
      for (int i = tid; i < 3 * HDIM; i += blockDim.x) iou_last[i] = s_gates[i];
    } else {
      for (int i = tid; i < 3 * HDIM; i += blockDim.x) iou_last[i] = seed_iou[i];
    }
    if (tid < HDIM) c_last[tid] = s_c[tid];
  }
}

// ---------------------------------------------------------------------------
// Kernel 4: per-node outputs: down/up scores + variable_chosen passthrough.
// ---------------------------------------------------------------------------
__global__ void finalize_node_kernel(const float* __restrict__ score,
                                     const float* __restrict__ sid,
                                     const float* __restrict__ siu,
                                     const int* __restrict__ vc,
                                     float* __restrict__ out_down,
                                     float* __restrict__ out_up,
                                     int* __restrict__ out_vc, int n) {
  int i = blockIdx.x * blockDim.x + threadIdx.x;
  if (i < n) {
    float s = score[i];
    out_down[i] = s * sid[i];
    out_up[i] = s * siu[i];
    out_vc[i] = vc[i];
  }
}

// ---------------------------------------------------------------------------
// Kernel 5: per-candidate masked mean + score blend. One block per candidate.
// ---------------------------------------------------------------------------
__global__ __launch_bounds__(256) void cand_kernel(
    const float* __restrict__ score, const float* __restrict__ sid,
    const float* __restrict__ siu, const int* __restrict__ vc,
    const int* __restrict__ cands, const float* __restrict__ gains,
    float* __restrict__ out_scores, float* __restrict__ ws_scores, int n) {
  int k = blockIdx.x;
  int cand = cands[k];
  int tid = threadIdx.x, lane = tid & 31, wave = tid >> 5;
  float sd = 0.0f, su = 0.0f, cnt = 0.0f;
  for (int i = tid; i < n; i += 256) {
    if (vc[i] == cand) {
      float s = score[i];
      sd += s * sid[i];
      su += s * siu[i];
      cnt += 1.0f;
    }
  }
#pragma unroll
  for (int off = 16; off > 0; off >>= 1) {
    sd += __shfl_xor(sd, off, 32);
    su += __shfl_xor(su, off, 32);
    cnt += __shfl_xor(cnt, off, 32);
  }
  __shared__ float rs[8], ru[8], rc[8];
  if (lane == 0) { rs[wave] = sd; ru[wave] = su; rc[wave] = cnt; }
  __syncthreads();
  if (tid == 0) {
    float S = 0.0f, U = 0.0f, C = 0.0f;
    for (int w = 0; w < 8; w++) { S += rs[w]; U += ru[w]; C += rc[w]; }
    float pd, pu;
    if (C > 0.0f) { pd = S / C; pu = U / C; }
    else { pd = gains[2 * k]; pu = gains[2 * k + 1]; }
    float scv = (1.0f - MU_C) * pd + MU_C * fmaxf(pd, pu);
    out_scores[k] = scv;
    ws_scores[k] = scv;
  }
}

__global__ void argmax_kernel(const float* __restrict__ ws_scores,
                              int* __restrict__ out_best, int kk) {
  if (blockIdx.x == 0 && threadIdx.x == 0) {
    int best = 0;
    float bv = ws_scores[0];
    for (int i = 1; i < kk; i++) {
      float v = ws_scores[i];
      if (v > bv) { bv = v; best = i; }
    }
    out_best[0] = best;
  }
}

// ---------------------------------------------------------------------------
extern "C" void kernel_launch(void* const* d_in, const int* in_sizes, int n_in,
                              void* d_out, int out_size, void* d_ws,
                              size_t ws_size, hipStream_t stream) {
  (void)n_in; (void)out_size; (void)ws_size;
  const float* feat   = (const float*)d_in[0];
  const float* c_in   = (const float*)d_in[2];
  const float* iou_in = (const float*)d_in[3];
  const float* sid    = (const float*)d_in[4];
  const float* siu    = (const float*)d_in[5];
  const int*   vc     = (const int*)d_in[6];
  const int*   cands  = (const int*)d_in[7];
  const float* gains  = (const float*)d_in[8];
  const float* W_iou  = (const float*)d_in[9];
  const float* W_f    = (const float*)d_in[10];
  const float* W_f_b  = (const float*)d_in[11];
  const float* U_iou  = (const float*)d_in[12];
  const float* U_f    = (const float*)d_in[13];
  const float* U_f_b  = (const float*)d_in[14];
  const float* b_iou  = (const float*)d_in[15];
  const float* b_f    = (const float*)d_in[16];
  const float* lin_w  = (const float*)d_in[17];
  const float* lin_b  = (const float*)d_in[18];

  int n  = in_sizes[4];  // N nodes
  int kk = in_sizes[7];  // K candidates
  if (n <= 0) return;

  // Carve workspace (256B aligned): ~232 MB total for N=100000.
  char* w = (char*)d_ws;
  auto carve = [&](size_t bytes) -> char* {
    char* p = w;
    w += (bytes + 255) & ~(size_t)255;
    return p;
  };
  __bf16* Wc        = (__bf16*)carve((size_t)GDIM * XDIM * 2);
  __bf16* Uc        = (__bf16*)carve((size_t)GDIM * HDIM * 2);
  float*  biasc     = (float*)carve((size_t)GDIM * 4);
  float*  iou_last  = (float*)carve((size_t)3 * HDIM * 4);
  float*  c_last    = (float*)carve((size_t)HDIM * 4);
  float*  ws_scores = (float*)carve((size_t)kk * 4);
  float*  score_nd  = (float*)carve((size_t)n * 4);
  __bf16* featb     = (__bf16*)carve((size_t)n * XDIM * 2);
  __bf16* pre       = (__bf16*)carve((size_t)n * GDIM * 2);

  // Phase 1: pack + GEMM (memory-bound, whole device).
  prep_kernel<<<512, 256, 0, stream>>>(feat, W_iou, W_f, U_iou, U_f, b_iou,
                                       W_f_b, U_f_b, b_f, Wc, Uc, biasc,
                                       featb, n);
  dim3 gg(GDIM / 128, (unsigned)((n + 63) / 64));
  pre_gemm_kernel<<<gg, 256, 0, stream>>>(featb, Wc, biasc, pre, n);

  // Phase 2: sequential chains (one WGP each, stream-ordered).
  chain_kernel<0><<<1, 1024, 0, stream>>>(pre, Uc, iou_in, c_in, iou_last,
                                          c_last, score_nd, lin_w, lin_b, n);
  chain_kernel<1><<<1, 1024, 0, stream>>>(pre, Uc, iou_last, c_last, iou_last,
                                          c_last, score_nd, lin_w, lin_b, n);

  // Phase 3: outputs. Layout: [best(int) | scores(kk) | down(n) | up(n) | vc(n)]
  float* out = (float*)d_out;
  int*   outi = (int*)d_out;
  float* out_scores = out + 1;
  float* out_down   = out + 1 + kk;
  float* out_up     = out + 1 + kk + (size_t)n;
  int*   out_vc     = outi + 1 + kk + 2 * (size_t)n;

  finalize_node_kernel<<<(n + 255) / 256, 256, 0, stream>>>(
      score_nd, sid, siu, vc, out_down, out_up, out_vc, n);
  cand_kernel<<<kk, 256, 0, stream>>>(score_nd, sid, siu, vc, cands, gains,
                                      out_scores, ws_scores, n);
  argmax_kernel<<<1, 64, 0, stream>>>(ws_scores, outi, kk);
}
